// MultiHeadAttention_79706003079680
// MI455X (gfx1250) — compile-verified, matrix operand order checked by reading, not run on hardware
//
#include <hip/hip_runtime.h>

// ---------------------------------------------------------------------------
// MI455X (gfx1250, wave32) multi-head-attention pipeline.
// All GEMMs via v_wmma_f32_16x16x32_bf16; global->LDS staging via the CDNA5
// async-to-LDS path (ASYNCcnt) with LDS double buffering; scores f32.
// ---------------------------------------------------------------------------

typedef __attribute__((ext_vector_type(16))) __bf16 v16bf;
typedef __attribute__((ext_vector_type(8)))  float  v8f;

#define BATCH 4
#define NCTX  2048
#define HID   512
#define PROJ  4096
#define BN    (BATCH * NCTX)      // 8192 total rows
#define ATT_SCALE 0.125f          // 64^-0.5

// ---- CDNA5 async global->LDS (guarded; falls back to batched sync staging) ----
#if defined(__AMDGCN__) && __has_builtin(__builtin_amdgcn_global_load_async_to_lds_b128)
#define USE_ASYNC_LDS 1
#else
#define USE_ASYNC_LDS 0
#endif

#if USE_ASYNC_LDS
// builtin signature wants GCC-vector int4 pointers: (as1 v4i*, as3 v4i*, imm, imm)
typedef int v4i_b128 __attribute__((vector_size(16)));
#define ASYNC_CP_B128(gptr, lptr)                                              \
  __builtin_amdgcn_global_load_async_to_lds_b128(                              \
      (__attribute__((address_space(1))) v4i_b128*)(gptr),                     \
      (__attribute__((address_space(3))) v4i_b128*)(lptr), 0, 0)
#if __has_builtin(__builtin_amdgcn_s_wait_asynccnt)
#define WAIT_ASYNC(n) __builtin_amdgcn_s_wait_asynccnt(n)
#else
#define WAIT_ASYNC(n) asm volatile("s_wait_asynccnt %0" ::"i"(n) : "memory")
#endif
#else
#define WAIT_ASYNC(n) do {} while (0)
#endif

// ---------------- conversion kernels ----------------
__global__ void k_cvt_bf16(const float* __restrict__ in, __bf16* __restrict__ out,
                           long long n) {
  long long i = (long long)blockIdx.x * blockDim.x + threadIdx.x;
  long long stride = (long long)gridDim.x * blockDim.x;
  for (; i < n; i += stride) out[i] = (__bf16)in[i];
}

// out[c*R + r] = bf16(in[r*C + c])   (weight transpose + downconvert)
__global__ void k_tcvt_bf16(const float* __restrict__ in, __bf16* __restrict__ out,
                            int R, int C) {
  long long n = (long long)R * C;
  long long i = (long long)blockIdx.x * blockDim.x + threadIdx.x;
  long long stride = (long long)gridDim.x * blockDim.x;
  for (; i < n; i += stride) {
    int r = (int)(i / C);
    int c = (int)(i % C);
    out[(size_t)c * R + r] = (__bf16)in[i];
  }
}

// ---------------- generic bf16 WMMA GEMM ----------------
// C[M,Nn] = epilogue( A[M,K] * Bt[Nn,K]^T ),  epilogue: (acc + bias[col]) * alpha
// outMode 0: f32 row-major   1: bf16 row-major
//         2: bf16 transposed per batch of transN rows: C[(b*Nn+col)*transN + r%transN]
#define TILE_M 128
#define TILE_N 128
#define TILE_K 32
#define LDSTR  40   // 32 + 8 pad bf16; 80B row pitch keeps 16B alignment
#define BUFE   (TILE_M * LDSTR)

__global__ __launch_bounds__(256) void k_gemm_bf16(
    const __bf16* __restrict__ A, long long sAb, int lda,
    const __bf16* __restrict__ Bt, long long sBb, int ldb,
    void* __restrict__ Cv, long long sCb, int ldc,
    int M, int Nn, int K, float alpha, const float* __restrict__ bias,
    int outMode, int transN) {
  __shared__ __bf16 lA[2 * BUFE];
  __shared__ __bf16 lB[2 * BUFE];

  const int tid  = threadIdx.x;
  const int wid  = tid >> 5;      // 8 waves
  const int lane = tid & 31;
  const int bz   = blockIdx.z;
  const int rowBase = blockIdx.y * TILE_M;
  const int colBase = blockIdx.x * TILE_N;

  A  += (size_t)bz * sAb;
  Bt += (size_t)bz * sBb;

  const int waveM = wid >> 2;     // 0..1 -> 64 rows each
  const int waveN = wid & 3;      // 0..3 -> 32 cols each

  v8f acc[4][2];
  v8f zeroV = {};
#pragma unroll
  for (int i = 0; i < 4; ++i)
#pragma unroll
    for (int j = 0; j < 2; ++j) acc[i][j] = zeroV;

  const int lm    = lane & 15;         // row/col within 16
  const int khalf = (lane >> 4) * 8;   // ISA bf16 A/B lane K pattern

  // stage one 128x32 A tile + 128x32 Bt tile into LDS buffer `buf`
  auto stage = [&](int k0, int buf) {
#if USE_ASYNC_LDS
#pragma unroll
    for (int i = 0; i < 2; ++i) {
      int linear = tid + i * 256;          // 0..511 chunks of 8 bf16
      int r = linear >> 2;                 // 0..127
      int c = (linear & 3) * 8;            // 0,8,16,24
      ASYNC_CP_B128(A + (size_t)(rowBase + r) * lda + k0 + c,
                    &lA[buf * BUFE + r * LDSTR + c]);
      ASYNC_CP_B128(Bt + (size_t)(colBase + r) * ldb + k0 + c,
                    &lB[buf * BUFE + r * LDSTR + c]);
    }
#else
    uint4 va[2], vb[2];
#pragma unroll
    for (int i = 0; i < 2; ++i) {
      int linear = tid + i * 256;
      int r = linear >> 2;
      int c = (linear & 3) * 8;
      va[i] = *(const uint4*)(A + (size_t)(rowBase + r) * lda + k0 + c);
      vb[i] = *(const uint4*)(Bt + (size_t)(colBase + r) * ldb + k0 + c);
    }
#pragma unroll
    for (int i = 0; i < 2; ++i) {
      int linear = tid + i * 256;
      int r = linear >> 2;
      int c = (linear & 3) * 8;
      *(uint4*)&lA[buf * BUFE + r * LDSTR + c] = va[i];
      *(uint4*)&lB[buf * BUFE + r * LDSTR + c] = vb[i];
    }
#endif
  };

  const int nTiles = K / TILE_K;
  stage(0, 0);

  for (int kt = 0; kt < nTiles; ++kt) {
    const int buf = kt & 1;
    // prefetch next tile into the other buffer while we compute this one
    if (kt + 1 < nTiles) {
      stage((kt + 1) * TILE_K, buf ^ 1);
      WAIT_ASYNC(4);   // 4 newest (next-tile) stay in flight; this tile landed
    } else {
      WAIT_ASYNC(0);
    }
    __syncthreads();   // all waves' stage(kt) visible

    // ---- fragments per the 16-bit A/B VGPR lane layout, then 8 WMMAs ----
    v16bf afr[4], bfr[2];
#pragma unroll
    for (int tm = 0; tm < 4; ++tm) {
      const __bf16* p = &lA[buf * BUFE + (waveM * 64 + tm * 16 + lm) * LDSTR];
      *(uint4*)&afr[tm]       = *(const uint4*)(p + khalf);       // K 0..7 / 8..15
      *((uint4*)&afr[tm] + 1) = *(const uint4*)(p + 16 + khalf);  // K 16..23 / 24..31
    }
#pragma unroll
    for (int tn = 0; tn < 2; ++tn) {
      const __bf16* p = &lB[buf * BUFE + (waveN * 32 + tn * 16 + lm) * LDSTR];
      *(uint4*)&bfr[tn]       = *(const uint4*)(p + khalf);
      *((uint4*)&bfr[tn] + 1) = *(const uint4*)(p + 16 + khalf);
    }
#pragma unroll
    for (int tm = 0; tm < 4; ++tm)
#pragma unroll
      for (int tn = 0; tn < 2; ++tn)
        acc[tm][tn] = __builtin_amdgcn_wmma_f32_16x16x32_bf16(
            false, afr[tm], false, bfr[tn], (short)0, acc[tm][tn], false, false);

    __syncthreads();   // everyone done reading buf before stage(kt+2) overwrites it
  }

  // ---- epilogue: C/D layout: VGPR r -> M=r (lanes 0-15) / M=8+r (16-31) ----
  const int nLocal = lane & 15;
  const int mHi    = (lane >> 4) * 8;
#pragma unroll
  for (int tm = 0; tm < 4; ++tm)
#pragma unroll
    for (int tn = 0; tn < 2; ++tn) {
      int colT = colBase + waveN * 32 + tn * 16 + nLocal;
      float bv = bias ? bias[colT] : 0.0f;
#pragma unroll
      for (int r = 0; r < 8; ++r) {
        int rowT = rowBase + waveM * 64 + tm * 16 + mHi + r;
        float val = (acc[tm][tn][r] + bv) * alpha;
        if (outMode == 0) {
          float* C = (float*)Cv + (size_t)bz * sCb;
          C[(size_t)rowT * ldc + colT] = val;
        } else if (outMode == 1) {
          __bf16* C = (__bf16*)Cv + (size_t)bz * sCb;
          C[(size_t)rowT * ldc + colT] = (__bf16)val;
        } else {
          __bf16* C = (__bf16*)Cv;
          int bb = rowT / transN, rr = rowT - bb * transN;
          C[((size_t)bb * Nn + colT) * (size_t)transN + rr] = (__bf16)val;
        }
      }
    }
}

// ---------------- softmax pass 1: per-row max & 1/sum ----------------
// S[b][m][n] is the transposed score matrix; reference softmax over query axis n
// == row softmax of S.  Coalesced row reads; stats to small scratch.
__global__ __launch_bounds__(256) void k_softmax_stats(
    const float* __restrict__ S, float* __restrict__ mxA, float* __restrict__ invA,
    int Nq, int Nk) {
  const int m = blockIdx.x;
  const int b = blockIdx.y;
  const float* row = S + ((size_t)b * Nk + m) * Nq;
  __shared__ float red[256];
  const int tid = threadIdx.x;

  float mx = -1e30f;
  for (int n = tid; n < Nq; n += 256) mx = fmaxf(mx, row[n]);
  red[tid] = mx;
  __syncthreads();
  for (int s = 128; s > 0; s >>= 1) {
    if (tid < s) red[tid] = fmaxf(red[tid], red[tid + s]);
    __syncthreads();
  }
  mx = red[0];
  __syncthreads();

  float sum = 0.0f;
  for (int n = tid; n < Nq; n += 256) sum += __expf(row[n] - mx);
  red[tid] = sum;
  __syncthreads();
  for (int s = 128; s > 0; s >>= 1) {
    if (tid < s) red[tid] += red[tid + s];
    __syncthreads();
  }
  if (tid == 0) {
    mxA[(size_t)b * Nk + m]  = mx;
    invA[(size_t)b * Nk + m] = 1.0f / red[0];
  }
}

// ---------------- softmax pass 2: exp-normalize + 64x64 LDS transpose ----------
// attn[b][n][m] = bf16( exp(S[b][m][n] - mx[m]) * inv[m] )
// Reads S coalesced (rows of m), writes attn coalesced (rows of n).
__global__ __launch_bounds__(256) void k_attn_transpose(
    const float* __restrict__ S, const float* __restrict__ mxA,
    const float* __restrict__ invA, __bf16* __restrict__ attn, int Nctx) {
  __shared__ float t[64][65];
  const int b  = blockIdx.z;
  const int m0 = blockIdx.y * 64;
  const int n0 = blockIdx.x * 64;
  const int tid = threadIdx.x;
  const int rr = tid >> 2;             // 0..63
  const int cc = (tid & 3) * 16;       // 0,16,32,48

  const int   mg  = m0 + rr;
  const float mx  = mxA[(size_t)b * Nctx + mg];
  const float inv = invA[(size_t)b * Nctx + mg];
  const float* srow = S + ((size_t)b * Nctx + mg) * Nctx + n0 + cc;
#pragma unroll
  for (int i = 0; i < 16; ++i)
    t[rr][cc + i] = __expf(srow[i] - mx) * inv;
  __syncthreads();

  __bf16* drow = attn + ((size_t)b * Nctx + (n0 + rr)) * Nctx + m0 + cc;
#pragma unroll
  for (int i = 0; i < 16; ++i)
    drow[i] = (__bf16)t[cc + i][rr];
}

// ---------------- launcher ----------------
extern "C" void kernel_launch(void* const* d_in, const int* in_sizes, int n_in,
                              void* d_out, int out_size, void* d_ws, size_t ws_size,
                              hipStream_t stream) {
  const float* q  = (const float*)d_in[0];
  const float* k  = (const float*)d_in[1];
  const float* v  = (const float*)d_in[2];
  const float* Wq = (const float*)d_in[3];
  const float* bq = (const float*)d_in[4];
  const float* Wk = (const float*)d_in[5];
  const float* bk = (const float*)d_in[6];
  const float* Wv = (const float*)d_in[7];
  const float* bv = (const float*)d_in[8];
  const float* Wo = (const float*)d_in[9];
  const float* bo = (const float*)d_in[10];
  float* out = (float*)d_out;

  char* ws = (char*)d_ws;
  size_t off = 0;
  auto alloc = [&](size_t bytes) -> size_t {
    size_t o = off;
    off += (bytes + 255) & ~(size_t)255;
    return o;
  };

  const size_t o_qbf  = alloc((size_t)BN * HID * 2);     // 8 MB
  const size_t o_kbf  = alloc((size_t)BN * HID * 2);
  const size_t o_vbf  = alloc((size_t)BN * HID * 2);
  const size_t o_WqT  = alloc((size_t)PROJ * HID * 2);   // 4 MB each
  const size_t o_WkT  = alloc((size_t)PROJ * HID * 2);
  const size_t o_WvT  = alloc((size_t)PROJ * HID * 2);
  const size_t o_WoT  = alloc((size_t)HID * PROJ * 2);
  const size_t o_qp   = alloc((size_t)BN * PROJ * 2);    // 64 MB (reused as attn)
  const size_t o_kp   = alloc((size_t)BN * PROJ * 2);    // 64 MB
  const size_t o_vpT  = alloc((size_t)BN * PROJ * 2);    // 64 MB
  const size_t o_S    = alloc((size_t)BATCH * NCTX * NCTX * 4); // 64 MB (reused as x)
  const size_t o_mx   = alloc((size_t)BATCH * NCTX * 4);
  const size_t o_inv  = alloc((size_t)BATCH * NCTX * 4);

  __bf16* qbf  = (__bf16*)(ws + o_qbf);
  __bf16* kbf  = (__bf16*)(ws + o_kbf);
  __bf16* vbf  = (__bf16*)(ws + o_vbf);
  __bf16* WqT  = (__bf16*)(ws + o_WqT);
  __bf16* WkT  = (__bf16*)(ws + o_WkT);
  __bf16* WvT  = (__bf16*)(ws + o_WvT);
  __bf16* WoT  = (__bf16*)(ws + o_WoT);
  __bf16* qp   = (__bf16*)(ws + o_qp);
  __bf16* kp   = (__bf16*)(ws + o_kp);
  __bf16* vpT  = (__bf16*)(ws + o_vpT);
  float*  S    = (float*)(ws + o_S);
  float*  mxA  = (float*)(ws + o_mx);
  float*  invA = (float*)(ws + o_inv);
  __bf16* attn = (__bf16*)(ws + o_qp);   // alias: qp dead after scores GEMM
  __bf16* xbf  = (__bf16*)(ws + o_S);    // alias: S dead after softmax pass 2

  const dim3 blk(256);

  // 1) inputs -> bf16; weights -> bf16 transposed (B-operand = row-major B^T)
  k_cvt_bf16<<<2048, blk, 0, stream>>>(q, qbf, (long long)BN * HID);
  k_cvt_bf16<<<2048, blk, 0, stream>>>(k, kbf, (long long)BN * HID);
  k_cvt_bf16<<<2048, blk, 0, stream>>>(v, vbf, (long long)BN * HID);
  k_tcvt_bf16<<<2048, blk, 0, stream>>>(Wq, WqT, HID, PROJ);
  k_tcvt_bf16<<<2048, blk, 0, stream>>>(Wk, WkT, HID, PROJ);
  k_tcvt_bf16<<<2048, blk, 0, stream>>>(Wv, WvT, HID, PROJ);
  k_tcvt_bf16<<<2048, blk, 0, stream>>>(Wo, WoT, PROJ, HID);

  // 2) projections: [8192,512] x [512,4096]
  k_gemm_bf16<<<dim3(PROJ / TILE_N, BN / TILE_M, 1), blk, 0, stream>>>(
      qbf, 0, HID, WqT, 0, HID, qp, 0, PROJ,
      BN, PROJ, HID, ATT_SCALE, bq, /*mode*/1, 0);
  k_gemm_bf16<<<dim3(PROJ / TILE_N, BN / TILE_M, 1), blk, 0, stream>>>(
      kbf, 0, HID, WkT, 0, HID, kp, 0, PROJ,
      BN, PROJ, HID, 1.0f, bk, /*mode*/1, 0);
  // vp stored transposed per batch: vpT[b][d][n]
  k_gemm_bf16<<<dim3(PROJ / TILE_N, BN / TILE_M, 1), blk, 0, stream>>>(
      vbf, 0, HID, WvT, 0, HID, vpT, 0, 0,
      BN, PROJ, HID, 1.0f, bv, /*mode*/2, NCTX);

  // 3) transposed scores: S[b][m][n] = kp[b][m] . qp[b][n], K = 4096 (f32)
  k_gemm_bf16<<<dim3(NCTX / TILE_N, NCTX / TILE_M, BATCH), blk, 0, stream>>>(
      kp, (long long)NCTX * PROJ, PROJ,
      qp, (long long)NCTX * PROJ, PROJ,
      S, (long long)NCTX * NCTX, NCTX,
      NCTX, NCTX, PROJ, 1.0f, nullptr, /*mode*/0, 0);

  // 4) softmax over query axis == row softmax of S (two passes, all coalesced)
  k_softmax_stats<<<dim3(NCTX, BATCH, 1), blk, 0, stream>>>(S, mxA, invA, NCTX, NCTX);
  k_attn_transpose<<<dim3(NCTX / 64, NCTX / 64, BATCH), blk, 0, stream>>>(
      S, mxA, invA, attn, NCTX);

  // 5) x[b][n][d] = sum_m attn[b][n][m] * vpT[b][d][m], K = 2048
  k_gemm_bf16<<<dim3(PROJ / TILE_N, NCTX / TILE_M, BATCH), blk, 0, stream>>>(
      attn, (long long)NCTX * NCTX, NCTX,
      vpT, (long long)PROJ * NCTX, NCTX,
      xbf, (long long)NCTX * PROJ, PROJ,
      NCTX, PROJ, NCTX, 1.0f, nullptr, /*mode*/1, 0);

  // 6) out = x @ Wo + bo : [8192,4096] x [4096,512] -> f32
  k_gemm_bf16<<<dim3(HID / TILE_N, BN / TILE_M, 1), blk, 0, stream>>>(
      xbf, 0, PROJ, WoT, 0, PROJ, out, 0, HID,
      BN, HID, PROJ, 1.0f, bo, /*mode*/0, 0);
}
